// GNN_QNetwork_89919435309559
// MI455X (gfx1250) — matrix operand order repaired and checked by reading, not compiled, not run on hardware
//
#include <hip/hip_runtime.h>
#include <hip/hip_bf16.h>

#define NNODES 100000
#define NEDGES 1600000
#define FDIM   128
#define HDIM   128
#define ADIM   8

typedef __attribute__((ext_vector_type(16))) __bf16 v16bf;
typedef __attribute__((ext_vector_type(8)))  float  v8f;

// ---------------------------------------------------------------- utilities
__global__ void fill_f32v4(float* __restrict__ p, float v, int n4) {
  int i = blockIdx.x * blockDim.x + threadIdx.x;
  if (i < n4) ((float4*)p)[i] = make_float4(v, v, v, v);
}

__global__ void deg_accum(const int* __restrict__ ei, float* __restrict__ deg) {
  int e = blockIdx.x * blockDim.x + threadIdx.x;
  if (e < NEDGES) atomicAdd(&deg[ei[NEDGES + e]], 1.0f);  // dst row
}

__global__ void deg_to_rsqrt(float* __restrict__ d) {
  int i = blockIdx.x * blockDim.x + threadIdx.x;
  if (i < NNODES) d[i] = rsqrtf(d[i]);
}

// ------------------------------------------------- WMMA GEMM: Y = act(X) @ W
// One 16x16 output tile per wave; 8 waves/block -> 16 rows x 128 cols/block.
// RELU applied during fp32->bf16 conversion of the A fragment (fuses the
// previous layer's activation for free).
template <bool RELU>
__global__ __launch_bounds__(256) void gemm128_bf16(const float* __restrict__ X,
                                                    const float* __restrict__ W,
                                                    float* __restrict__ Y) {
  __shared__ __bf16 wt[HDIM][HDIM + 8];  // wt[n][k] = W[k][n], padded
  const int tid = threadIdx.x;
  for (int i = tid; i < HDIM * HDIM; i += 256) {
    int k = i >> 7, n = i & 127;
    wt[n][k] = (__bf16)W[i];
  }
  __syncthreads();

  const int wave = tid >> 5, lane = tid & 31;
  const int g = lane >> 4, l16 = lane & 15;
  const int row0 = blockIdx.x * 16;          // N % 16 == 0, no edge handling
  const int row  = row0 + l16;
  const int col  = wave * 16 + l16;

  const float* xr = X + (size_t)row * FDIM;
  v8f acc = {};
#pragma unroll
  for (int kb = 0; kb < 4; ++kb) {
    const int kbase = kb * 32;
    v16bf a, b;
#pragma unroll
    for (int j = 0; j < 8; ++j) {            // A 16x32: contig K = 16*g + s
      float lo = xr[kbase + g * 16 + j];
      float hi = xr[kbase + g * 16 + 8 + j];
      if (RELU) { lo = fmaxf(lo, 0.f); hi = fmaxf(hi, 0.f); }
      a[j]     = (__bf16)lo;
      a[8 + j] = (__bf16)hi;
    }
    const __bf16* wr = &wt[col][kbase + g * 16];  // B 32x16: lane=col, contig K
#pragma unroll
    for (int j = 0; j < 16; ++j) b[j] = wr[j];
    acc = __builtin_amdgcn_wmma_f32_16x16x32_bf16(false, a, false, b,
                                                  (short)0, acc, false, false);
  }
#pragma unroll
  for (int v = 0; v < 8; ++v)                // C/D: M = v + 8*g, N = lane&15
    Y[(size_t)(row0 + v + 8 * g) * HDIM + col] = acc[v];
}

// ---------------- agg = hw*dis^2 + b  (self-loop + bias fused, replaces fill)
__global__ __launch_bounds__(256) void agg_init(const float* __restrict__ hw,
                                                const float* __restrict__ dis,
                                                const float* __restrict__ b,
                                                float* __restrict__ agg) {
  const long long q = (long long)blockIdx.x * 256 + threadIdx.x;  // float4 idx
  const int i = (int)(q >> 5);          // node  (32 float4 per row)
  const int j4 = (int)(q & 31);         // float4 within row
  float s = dis[i]; s *= s;
  const float4 h  = ((const float4*)hw)[q];
  const float4 bb = ((const float4*)b)[j4];
  float4 r;
  r.x = h.x * s + bb.x;
  r.y = h.y * s + bb.y;
  r.z = h.z * s + bb.z;
  r.w = h.w * s + bb.w;
  ((float4*)agg)[q] = r;
}

// ------------------------------------------------- edge scatter: wave per edge
__global__ __launch_bounds__(256) void edge_agg(const int* __restrict__ ei,
                                                const float* __restrict__ dis,
                                                const float* __restrict__ hw,
                                                float* __restrict__ agg) {
  const int e = blockIdx.x * 8 + (threadIdx.x >> 5);
  if (e >= NEDGES) return;
  const int lane = threadIdx.x & 31;
  const int s = ei[e], d = ei[NEDGES + e];
  const float w = dis[s] * dis[d];
  const float4 m = ((const float4*)(hw + (size_t)s * HDIM))[lane];
  float* ap = agg + (size_t)d * HDIM + lane * 4;
  atomicAdd(ap + 0, m.x * w);
  atomicAdd(ap + 1, m.y * w);
  atomicAdd(ap + 2, m.z * w);
  atomicAdd(ap + 3, m.w * w);
}

// ------------------------------------------------- FC head: out = relu(H) @ Wfc + bfc
// Wfc is 128x8, zero-padded to 16 cols in LDS. One 16-row tile per wave.
__global__ __launch_bounds__(256) void gemm_fc(const float* __restrict__ Hm,
                                               const float* __restrict__ Wfc,
                                               const float* __restrict__ bfc,
                                               float* __restrict__ out) {
  __shared__ __bf16 wt[16][HDIM + 8];  // wt[n][k] = (n<8) ? Wfc[k][n] : 0
  const int tid = threadIdx.x;
  for (int i = tid; i < 16 * HDIM; i += 256) {
    int n = i >> 7, k = i & 127;
    wt[n][k] = (n < ADIM) ? (__bf16)Wfc[k * ADIM + n] : (__bf16)0.f;
  }
  __syncthreads();

  const int wave = tid >> 5, lane = tid & 31;
  const int g = lane >> 4, l16 = lane & 15;
  const int row0 = blockIdx.x * 128 + wave * 16;
  const int row  = row0 + l16;
  const bool rowok = row0 + 15 < NNODES;   // tiles fully in or fully out

  const float* xr = Hm + (size_t)row * HDIM;
  v8f acc = {};
#pragma unroll
  for (int kb = 0; kb < 4; ++kb) {
    const int kbase = kb * 32;
    v16bf a = {}, b;
    if (rowok) {
#pragma unroll
      for (int j = 0; j < 8; ++j) {
        a[j]     = (__bf16)fmaxf(xr[kbase + g * 16 + j], 0.f);
        a[8 + j] = (__bf16)fmaxf(xr[kbase + g * 16 + 8 + j], 0.f);
      }
    }
    const __bf16* wr = &wt[l16][kbase + g * 16];
#pragma unroll
    for (int j = 0; j < 16; ++j) b[j] = wr[j];
    acc = __builtin_amdgcn_wmma_f32_16x16x32_bf16(false, a, false, b,
                                                  (short)0, acc, false, false);
  }
  if (rowok && l16 < ADIM) {
    const float bias = bfc[l16];
#pragma unroll
    for (int v = 0; v < 8; ++v)
      out[(size_t)(row0 + v + 8 * g) * ADIM + l16] = acc[v] + bias;
  }
}

// ---------------------------------------------------------------- launcher
extern "C" void kernel_launch(void* const* d_in, const int* in_sizes, int n_in,
                              void* d_out, int out_size, void* d_ws, size_t ws_size,
                              hipStream_t stream) {
  const float* x   = (const float*)d_in[0];
  const int*   ei  = (const int*)  d_in[1];
  const float* W1  = (const float*)d_in[2];
  const float* b1  = (const float*)d_in[3];
  const float* W2  = (const float*)d_in[4];
  const float* b2  = (const float*)d_in[5];
  const float* Wfc = (const float*)d_in[6];
  const float* bfc = (const float*)d_in[7];
  float* out = (float*)d_out;

  const size_t NH = (size_t)NNODES * HDIM;   // 12.8M floats
  float* buf0 = (float*)d_ws;                // hw buffer
  float* buf1 = buf0 + NH;                   // agg / h buffer
  float* deg  = buf1 + NH;                   // degree -> dis (N floats)

  const int nh4Blocks = (int)(NH / 4 / 256);  // 12500

  // degree + symmetric norm
  fill_f32v4<<<(NNODES / 4 + 255) / 256, 256, 0, stream>>>(deg, 1.0f, NNODES / 4);
  deg_accum<<<(NEDGES + 255) / 256, 256, 0, stream>>>(ei, deg);
  deg_to_rsqrt<<<(NNODES + 255) / 256, 256, 0, stream>>>(deg);

  // ---- layer 1: hw1 = x@W1 ; agg1 = hw1*dis^2 + b1 + scatter
  gemm128_bf16<false><<<NNODES / 16, 256, 0, stream>>>(x, W1, buf0);
  agg_init<<<nh4Blocks, 256, 0, stream>>>(buf0, deg, b1, buf1);
  edge_agg<<<NEDGES / 8, 256, 0, stream>>>(ei, deg, buf0, buf1);

  // ---- layer 2: hw2 = relu(agg1)@W2 (relu fused into A load)
  gemm128_bf16<true><<<NNODES / 16, 256, 0, stream>>>(buf1, W2, buf0);
  agg_init<<<nh4Blocks, 256, 0, stream>>>(buf0, deg, b2, buf1);
  edge_agg<<<NEDGES / 8, 256, 0, stream>>>(ei, deg, buf0, buf1);

  // ---- FC head: out = relu(agg2)@Wfc + bfc (relu fused into A load)
  gemm_fc<<<(NNODES + 127) / 128, 256, 0, stream>>>(buf1, Wfc, bfc, out);
}
